// MPTAttention_71184787964190
// MI455X (gfx1250) — matrix operand order repaired and checked by reading, not compiled, not run on hardware
//
#include <hip/hip_runtime.h>
#include <hip/hip_bf16.h>
#include <math.h>

#define B_SZ 2
#define S_LEN 2048
#define NH 32
#define HDIM 128
#define DM 4096
#define TDM 12288 /* 3*DM */
#define CLIPV 8.0f

typedef __attribute__((ext_vector_type(16))) _Float16 v16h;
typedef __attribute__((ext_vector_type(8)))  _Float16 v8h;
typedef __attribute__((ext_vector_type(4)))  _Float16 v4h;
typedef __attribute__((ext_vector_type(8)))  float    v8f;
typedef __attribute__((ext_vector_type(4)))  float    v4f32;

static __device__ __forceinline__ v16h cat8(v8h lo, v8h hi) {
  return __builtin_shufflevector(lo, hi, 0,1,2,3,4,5,6,7,8,9,10,11,12,13,14,15);
}
static __device__ __forceinline__ v8f wmma_f16(v16h a, v16h b, v8f c) {
  // D(16x16 f32) = A(16x32 f16) * B(32x16 f16) + C
  return __builtin_amdgcn_wmma_f32_16x16x32_f16(false, a, false, b, (short)0, c, false, false);
}
// CDNA5 async DMA: global -> LDS, tracked by ASYNCcnt (bypasses VGPRs).
static __device__ __forceinline__ void async_ld_b128(unsigned lds_addr, const void* gaddr) {
  asm volatile("global_load_async_to_lds_b128 %0, %1, off" :: "v"(lds_addr), "v"(gaddr) : "memory");
}
static __device__ __forceinline__ void wait_async_le4() {
  asm volatile("s_wait_asynccnt 4" ::: "memory");
}
static __device__ __forceinline__ void wait_async_0() {
  asm volatile("s_wait_asynccnt 0" ::: "memory");
}

// ---------------------------------------------------------------- fp32 -> f16
__global__ __launch_bounds__(256) void cvt_f16_kernel(const float* __restrict__ in,
                                                      _Float16* __restrict__ out, int n4) {
  int i = blockIdx.x * 256 + threadIdx.x;
  if (i < n4) {
    v4f32 x = *(const v4f32*)(in + (size_t)i * 4);
    v4h y;
    y[0] = (_Float16)x[0]; y[1] = (_Float16)x[1];
    y[2] = (_Float16)x[2]; y[3] = (_Float16)x[3];
    *(v4h*)(out + (size_t)i * 4) = y;
  }
}

// ---------------------------------------------------------------- NT GEMM
// C[m,n] = sum_k A[m,k] * Bm[n,k];  A:[M][K], Bm:[N][K] f16; K%32==0,
// M%128==0, N%128==0.  Block = 256 thr = 8 waves (4 M-waves x 2 N-waves),
// block tile 128x128, wave tile 32x64 (2x4 WMMA), K step 32.
// Global->LDS staging uses CDNA5 async-DMA loads with double buffering.
template <bool F16OUT>
__global__ __launch_bounds__(256) void gemm_nt_kernel(const _Float16* __restrict__ A,
                                                      const _Float16* __restrict__ Bm,
                                                      float* __restrict__ Cf,
                                                      _Float16* __restrict__ Ch,
                                                      int M, int N, int K, float clipv) {
  __shared__ __align__(16) _Float16 As[2][128 * 40]; // padded stride 40 halves
  __shared__ __align__(16) _Float16 Bs[2][128 * 40];

  const int t    = threadIdx.x;
  const int lane = t & 31;
  const int wave = t >> 5;
  const int wm   = wave >> 1;      // 0..3 -> 32 rows
  const int wn   = wave & 1;       // 0..1 -> 64 cols
  const int half = lane >> 4;
  const int l15  = lane & 15;
  const int bm   = blockIdx.y * 128;
  const int bn   = blockIdx.x * 128;

  v8f acc[2][4];
#pragma unroll
  for (int mt = 0; mt < 2; ++mt)
#pragma unroll
    for (int nt = 0; nt < 4; ++nt)
#pragma unroll
      for (int r = 0; r < 8; ++r) acc[mt][nt][r] = 0.f;

  // async staging: thread t moves 32B of A and 32B of B per K-step
  const int row = t >> 1;                   // 0..127
  const int seg = (t & 1) * 16;             // halves
  const _Float16* agp = A  + (size_t)(bm + row) * K + seg;
  const _Float16* bgp = Bm + (size_t)(bn + row) * K + seg;
  const unsigned lrow = (unsigned)((row * 40 + seg) * 2); // bytes in LDS tile
  const unsigned asOff = (unsigned)(uintptr_t)(&As[0][0]) + lrow;
  const unsigned bsOff = (unsigned)(uintptr_t)(&Bs[0][0]) + lrow;
  const unsigned bufB  = 128u * 40u * 2u;   // buffer stride (bytes)

  const int steps = K >> 5;
  // prime step 0
  async_ld_b128(asOff,      agp);
  async_ld_b128(asOff + 16, (const char*)agp + 16);
  async_ld_b128(bsOff,      bgp);
  async_ld_b128(bsOff + 16, (const char*)bgp + 16);

  for (int j = 0; j < steps; ++j) {
    if (j + 1 < steps) {                    // issue DMA for next step, then
      const int kb = (j + 1) << 5;          // drain the 4 oldest (this step)
      const unsigned bo = (unsigned)((j + 1) & 1) * bufB;
      async_ld_b128(asOff + bo,      agp + kb);
      async_ld_b128(asOff + bo + 16, (const char*)(agp + kb) + 16);
      async_ld_b128(bsOff + bo,      bgp + kb);
      async_ld_b128(bsOff + bo + 16, (const char*)(bgp + kb) + 16);
      wait_async_le4();
    } else {
      wait_async_0();
    }
    __syncthreads();

    const _Float16* Ab = As[j & 1];
    const _Float16* Bb = Bs[j & 1];
    v16h af[2], bf[4];
#pragma unroll
    for (int mt = 0; mt < 2; ++mt) {
      // A frag: lane row = l15, elements K = half*8 + (0..7 | 16..23)
      const _Float16* p = &Ab[(wm * 32 + mt * 16 + l15) * 40 + half * 8];
      af[mt] = cat8(*(const v8h*)p, *(const v8h*)(p + 16));
    }
#pragma unroll
    for (int nt = 0; nt < 4; ++nt) {
      // B frag: lane col = l15, elements K = half*8 + (0..7 | 16..23)
      const _Float16* p = &Bb[(wn * 64 + nt * 16 + l15) * 40 + half * 8];
      bf[nt] = cat8(*(const v8h*)p, *(const v8h*)(p + 16));
    }
#pragma unroll
    for (int mt = 0; mt < 2; ++mt)
#pragma unroll
      for (int nt = 0; nt < 4; ++nt)
        acc[mt][nt] = wmma_f16(af[mt], bf[nt], acc[mt][nt]);
    __syncthreads();
  }

#pragma unroll
  for (int mt = 0; mt < 2; ++mt)
#pragma unroll
    for (int nt = 0; nt < 4; ++nt)
#pragma unroll
      for (int r = 0; r < 8; ++r) {
        const int orow = bm + wm * 32 + mt * 16 + half * 8 + r;
        const int ocol = bn + wn * 64 + nt * 16 + l15;
        float v = acc[mt][nt][r];
        if (F16OUT) {
          v = fminf(fmaxf(v, -clipv), clipv);
          Ch[(size_t)orow * N + ocol] = (_Float16)v;
        } else {
          Cf[(size_t)orow * N + ocol] = v;
        }
      }
}

// ---------------------------------------------------------------- LayerNorm (in place on q or k third)
__global__ __launch_bounds__(256) void ln_kernel(_Float16* __restrict__ qkv,
                                                 const float* __restrict__ w,
                                                 const float* __restrict__ bvec, int part) {
  const int tid = threadIdx.x;
  _Float16* row = qkv + (size_t)blockIdx.x * TDM + part * DM;
  v8h xa = *(const v8h*)(row + tid * 16);
  v8h xb = *(const v8h*)(row + tid * 16 + 8);
  float x[16], s = 0.f, ss = 0.f;
#pragma unroll
  for (int i = 0; i < 8; ++i) { x[i] = (float)xa[i]; x[8 + i] = (float)xb[i]; }
#pragma unroll
  for (int i = 0; i < 16; ++i) { s += x[i]; ss += x[i] * x[i]; }
#pragma unroll
  for (int m = 16; m >= 1; m >>= 1) { s += __shfl_xor(s, m, 32); ss += __shfl_xor(ss, m, 32); }
  __shared__ float redS[8], redQ[8];
  if ((tid & 31) == 0) { redS[tid >> 5] = s; redQ[tid >> 5] = ss; }
  __syncthreads();
  s = 0.f; ss = 0.f;
#pragma unroll
  for (int i = 0; i < 8; ++i) { s += redS[i]; ss += redQ[i]; }
  const float mean = s * (1.0f / DM);
  const float var  = ss * (1.0f / DM) - mean * mean;
  const float rstd = rsqrtf(var + 1e-5f);
  v8h ya, yb;
#pragma unroll
  for (int i = 0; i < 8; ++i) {
    const int c0 = tid * 16 + i, c1 = tid * 16 + 8 + i;
    ya[i] = (_Float16)((x[i]     - mean) * rstd * w[c0] + bvec[c0]);
    yb[i] = (_Float16)((x[8 + i] - mean) * rstd * w[c1] + bvec[c1]);
  }
  *(v8h*)(row + tid * 16)     = ya;
  *(v8h*)(row + tid * 16 + 8) = yb;
}

// ---------------------------------------------------------------- V transpose: qkv v-part -> vT[B][H][Dh][S]
__global__ __launch_bounds__(256) void vtrans_kernel(const _Float16* __restrict__ qkv,
                                                     _Float16* __restrict__ vT) {
  __shared__ _Float16 tile[32][33];
  const int bh = blockIdx.z;            // b*NH + h
  const int b  = bh >> 5, h = bh & 31;
  const int s0 = blockIdx.x * 32, d0 = blockIdx.y * 32;
  const int tid = threadIdx.x;
#pragma unroll
  for (int i = 0; i < 4; ++i) {
    const int idx = tid + i * 256;
    const int sr = idx >> 5, dc = idx & 31;
    tile[sr][dc] = qkv[(size_t)(b * S_LEN + s0 + sr) * TDM + 2 * DM + h * HDIM + d0 + dc];
  }
  __syncthreads();
#pragma unroll
  for (int i = 0; i < 4; ++i) {
    const int idx = tid + i * 256;
    const int dr = idx >> 5, sc = idx & 31;
    vT[((size_t)bh * HDIM + d0 + dr) * S_LEN + s0 + sc] = tile[sc][dr];
  }
}

// ---------------------------------------------------------------- Flash attention (transposed-score trick)
// One wave per (b, h, 16-row q tile).  S^T = K * Q^T so both operands are
// d-contiguous; the S^T C-layout is bit-identical to the A-fragment layout
// needed by P for the PV WMMA, so no cross-lane traffic for P.
__global__ __launch_bounds__(256) void attn_kernel(const _Float16* __restrict__ qkv,
                                                   const _Float16* __restrict__ vT,
                                                   _Float16* __restrict__ ctx) {
  const int lane = threadIdx.x & 31;
  const int wave = threadIdx.x >> 5;
  const int gw = blockIdx.x * 8 + wave;   // B*NH*(S/16) waves total
  const int qt = gw & 127;
  const int h  = (gw >> 7) & 31;
  const int b  = gw >> 12;
  const int q0 = qt << 4;
  const int half = lane >> 4;
  const int l15  = lane & 15;
  const int qi   = q0 + l15;              // column this lane owns in S^T

  const float scale = 0.08838834764831845f;          // 1/sqrt(128)
  const float slope = exp2f(-0.25f * (float)(h + 1)); // ALiBi (H=32 power of 2)

  // Q B-fragments (persistent): lane col = qi, elements d = dc*32 + half*8 + (0..7 | 16..23)
  const _Float16* qrow = qkv + (size_t)(b * S_LEN + q0 + l15) * TDM + h * HDIM;
  v16h qf[4];
#pragma unroll
  for (int dc = 0; dc < 4; ++dc) {
    const _Float16* p = qrow + dc * 32 + half * 8;
    qf[dc] = cat8(*(const v8h*)p, *(const v8h*)(p + 16));
  }

  v8f oacc[8];
#pragma unroll
  for (int dt = 0; dt < 8; ++dt)
#pragma unroll
    for (int r = 0; r < 8; ++r) oacc[dt][r] = 0.f;

  float mrow = -1.0e30f, lrow = 0.f;
  const _Float16* kbase = qkv + (size_t)(b * S_LEN) * TDM + DM + h * HDIM;
  const _Float16* vbase = vT + (size_t)((b * NH + h) * HDIM) * S_LEN;

  for (int kc = 0; kc <= q0 + 15; kc += 32) {
    // ---- S^T tiles: rows kj (two 16-tiles), cols qi
    v8f s0, s1;
#pragma unroll
    for (int r = 0; r < 8; ++r) { s0[r] = 0.f; s1[r] = 0.f; }
#pragma unroll
    for (int dc = 0; dc < 4; ++dc) {
      const _Float16* kp0 = kbase + (size_t)(kc + l15) * TDM + dc * 32 + half * 8;
      const _Float16* kp1 = kbase + (size_t)(kc + 16 + l15) * TDM + dc * 32 + half * 8;
      v16h ka0 = cat8(*(const v8h*)kp0, *(const v8h*)(kp0 + 16));
      v16h ka1 = cat8(*(const v8h*)kp1, *(const v8h*)(kp1 + 16));
      s0 = wmma_f16(ka0, qf[dc], s0);
      s1 = wmma_f16(ka1, qf[dc], s1);
    }
    // ---- scale + ALiBi + causal mask, online softmax (per-lane column)
    float p0[8], p1[8];
    float cmax = -1.0e30f;
#pragma unroll
    for (int r = 0; r < 8; ++r) {
      const int kj0 = kc + half * 8 + r;
      const int kj1 = kj0 + 16;
      float a = s0[r] * scale + slope * (float)(kj0 - qi);
      float c = s1[r] * scale + slope * (float)(kj1 - qi);
      a = (kj0 <= qi) ? a : -1.0e30f;
      c = (kj1 <= qi) ? c : -1.0e30f;
      p0[r] = a; p1[r] = c;
      cmax = fmaxf(cmax, fmaxf(a, c));
    }
    cmax = fmaxf(cmax, __shfl_xor(cmax, 16, 32));   // join kj halves of this column
    const float mnew = fmaxf(mrow, cmax);
    float csum = 0.f;
#pragma unroll
    for (int r = 0; r < 8; ++r) {
      p0[r] = __expf(p0[r] - mnew);
      p1[r] = __expf(p1[r] - mnew);
      csum += p0[r] + p1[r];
    }
    csum += __shfl_xor(csum, 16, 32);
    const float alpha = __expf(mrow - mnew);
    lrow = lrow * alpha + csum;
    mrow = mnew;
    // ---- rescale O rows (broadcast alpha from S^T lane layout to O row layout)
#pragma unroll
    for (int r = 0; r < 8; ++r) {
      const float ar = __shfl(alpha, half * 8 + r, 32);
#pragma unroll
      for (int dt = 0; dt < 8; ++dt) oacc[dt][r] *= ar;
    }
    // ---- P fragment: S^T C-layout == P A-layout, just convert in place
    v16h pa;
#pragma unroll
    for (int r = 0; r < 8; ++r) { pa[r] = (_Float16)p0[r]; pa[8 + r] = (_Float16)p1[r]; }
    // ---- O += P * V  (V^T buffer makes B elements kj-contiguous)
#pragma unroll
    for (int dt = 0; dt < 8; ++dt) {
      const _Float16* vp = vbase + (size_t)(dt * 16 + l15) * S_LEN + kc + half * 8;
      v16h vb = cat8(*(const v8h*)vp, *(const v8h*)(vp + 16));
      oacc[dt] = wmma_f16(pa, vb, oacc[dt]);
    }
  }

  // ---- normalize and store
  _Float16* crow = ctx + (size_t)(b * S_LEN + q0) * DM + h * HDIM;
#pragma unroll
  for (int r = 0; r < 8; ++r) {
    const float lr = __shfl(lrow, half * 8 + r, 32);
    const float linv = 1.0f / lr;
#pragma unroll
    for (int dt = 0; dt < 8; ++dt)
      crow[(size_t)(half * 8 + r) * DM + dt * 16 + l15] = (_Float16)(oacc[dt][r] * linv);
  }
}

// ---------------------------------------------------------------- launch
extern "C" void kernel_launch(void* const* d_in, const int* in_sizes, int n_in,
                              void* d_out, int out_size, void* d_ws, size_t ws_size,
                              hipStream_t stream) {
  (void)in_sizes; (void)n_in; (void)out_size; (void)ws_size;
  const float* hidden = (const float*)d_in[0];
  const float* w_qkv  = (const float*)d_in[1];
  const float* q_ln_w = (const float*)d_in[2];
  const float* q_ln_b = (const float*)d_in[3];
  const float* k_ln_w = (const float*)d_in[4];
  const float* k_ln_b = (const float*)d_in[5];
  const float* w_out  = (const float*)d_in[6];
  float* out = (float*)d_out;

  const int M = B_SZ * S_LEN;  // 4096 rows
  _Float16* h16    = (_Float16*)d_ws;                       // M*DM
  _Float16* wqkv16 = h16    + (size_t)M * DM;               // TDM*DM
  _Float16* wout16 = wqkv16 + (size_t)TDM * DM;             // DM*DM
  _Float16* qkv16  = wout16 + (size_t)DM * DM;              // M*TDM
  _Float16* vTbuf  = qkv16  + (size_t)M * TDM;              // M*DM
  _Float16* ctx16  = vTbuf  + (size_t)M * DM;               // M*DM

  cvt_f16_kernel<<<(M * DM) / 1024,   256, 0, stream>>>(hidden, h16,    (M * DM) / 4);
  cvt_f16_kernel<<<(TDM * DM) / 1024, 256, 0, stream>>>(w_qkv, wqkv16, (TDM * DM) / 4);
  cvt_f16_kernel<<<(DM * DM) / 1024,  256, 0, stream>>>(w_out, wout16, (DM * DM) / 4);

  dim3 g1(TDM / 128, M / 128);
  gemm_nt_kernel<true><<<g1, 256, 0, stream>>>(h16, wqkv16, nullptr, qkv16, M, TDM, DM, CLIPV);

  ln_kernel<<<M, 256, 0, stream>>>(qkv16, q_ln_w, q_ln_b, 0);
  ln_kernel<<<M, 256, 0, stream>>>(qkv16, k_ln_w, k_ln_b, 1);

  dim3 gt(S_LEN / 32, HDIM / 32, B_SZ * NH);
  vtrans_kernel<<<gt, 256, 0, stream>>>(qkv16, vTbuf);

  attn_kernel<<<(B_SZ * NH * (S_LEN / 16)) / 8, 256, 0, stream>>>(qkv16, vTbuf, ctx16);

  dim3 g2(DM / 128, M / 128);
  gemm_nt_kernel<false><<<g2, 256, 0, stream>>>(ctx16, wout16, out, nullptr, M, DM, DM, 0.f);
}
